// GraphZoomNeuralNetwork_48120813584664
// MI455X (gfx1250) — matrix operand order repaired and checked by reading, not compiled, not run on hardware
//
#include <hip/hip_runtime.h>
#include <hip/hip_bf16.h>

// ---------------------------------------------------------------------------
// Problem constants (from the reference)
// ---------------------------------------------------------------------------
#define B_  512
#define T_  112
#define D_  1024
#define H_  1024
#define O_  128
#define LN_ 5
#define HO_ (LN_ * H_)   // 5120

typedef __attribute__((ext_vector_type(16))) __bf16 bf16x16;
typedef __attribute__((ext_vector_type(8)))  __bf16 bf16x8;
typedef __attribute__((ext_vector_type(8)))  float  f32x8;

// ---------------------------------------------------------------------------
// Device helpers
// ---------------------------------------------------------------------------
__device__ __forceinline__ __bf16 to_bf16(float f) { return (__bf16)f; }

// A fragment: 16x32 bf16, row-major source A (lda = K).
// Lane L: row = m0 + (L&15); 16 contiguous K [kbase, kbase+16),
// kbase = kb*32 + (L>=16 ? 16 : 0).   (contiguous k order, same as load_b_frag)
__device__ __forceinline__ bf16x16 load_a_frag(const __bf16* __restrict__ A, int lda,
                                               int m0, int kb, int lane) {
  const int row   = m0 + (lane & 15);
  const int kbase = (kb << 5) + ((lane & 16) ? 16 : 0);
  const __bf16* p = A + (size_t)row * lda + kbase;
  bf16x8 lo = *(const bf16x8*)(p);
  bf16x8 hi = *(const bf16x8*)(p + 8);
  bf16x16 r;
#pragma unroll
  for (int i = 0; i < 8; ++i) { r[i] = lo[i]; r[i + 8] = hi[i]; }
  return r;
}

// B fragment: 32x16 bf16 from pre-transposed weights BT (N x K row-major, ldb = K).
// Lane L: col = n0 + (L&15); 16 contiguous K starting at kb*32 + (L>=16 ? 16 : 0).
__device__ __forceinline__ bf16x16 load_b_frag(const __bf16* __restrict__ BT, int ldb,
                                               int n0, int kb, int lane) {
  const int col   = n0 + (lane & 15);
  const int kbase = (kb << 5) + ((lane & 16) ? 16 : 0);
  return *(const bf16x16*)(BT + (size_t)col * ldb + kbase);
}

__device__ __forceinline__ f32x8 wmma_bf16(bf16x16 a, bf16x16 b, f32x8 c) {
  return __builtin_amdgcn_wmma_f32_16x16x32_bf16(false, a, false, b, (short)0, c,
                                                 false, false);
}

// ---------------------------------------------------------------------------
// Fused multi-term GEMM:  C = act( sum_i A_i * B_i^T + bias )
// Each wave computes a 16(M) x 64(N) strip; fp32 accumulate in v8f x4.
// ---------------------------------------------------------------------------
template <int NT, bool TANH>
__launch_bounds__(256)
__global__ void gemm_bf16_kernel(const __bf16* __restrict__ A1, const __bf16* __restrict__ B1,
                                 const __bf16* __restrict__ A2, const __bf16* __restrict__ B2,
                                 const __bf16* __restrict__ A3, const __bf16* __restrict__ B3,
                                 const float* __restrict__ bias, float* __restrict__ C,
                                 int M, int N, int K) {
  const int lane   = threadIdx.x & 31;
  const int wid    = (blockIdx.x * blockDim.x + threadIdx.x) >> 5;
  const int mtiles = M >> 4;
  const int ntiles = N >> 6;
  if (wid >= mtiles * ntiles) return;
  const int m0 = (wid % mtiles) << 4;
  const int n0 = (wid / mtiles) << 6;

  f32x8 acc[4] = {f32x8{}, f32x8{}, f32x8{}, f32x8{}};
  const int kblocks = K >> 5;
  for (int kb = 0; kb < kblocks; ++kb) {
    {
      bf16x16 a = load_a_frag(A1, K, m0, kb, lane);
#pragma unroll
      for (int j = 0; j < 4; ++j)
        acc[j] = wmma_bf16(a, load_b_frag(B1, K, n0 + (j << 4), kb, lane), acc[j]);
    }
    if (NT > 1) {
      bf16x16 a = load_a_frag(A2, K, m0, kb, lane);
#pragma unroll
      for (int j = 0; j < 4; ++j)
        acc[j] = wmma_bf16(a, load_b_frag(B2, K, n0 + (j << 4), kb, lane), acc[j]);
    }
    if (NT > 2) {
      bf16x16 a = load_a_frag(A3, K, m0, kb, lane);
#pragma unroll
      for (int j = 0; j < 4; ++j)
        acc[j] = wmma_bf16(a, load_b_frag(B3, K, n0 + (j << 4), kb, lane), acc[j]);
    }
  }

  // C/D layout: VGPR r -> row m0 + r + (lane>=16 ? 8 : 0), col n0 + 16j + (lane&15)
  const int roff = (lane & 16) ? 8 : 0;
  const int cl   = lane & 15;
#pragma unroll
  for (int j = 0; j < 4; ++j) {
    const int col  = n0 + (j << 4) + cl;
    const float bv = bias ? bias[col] : 0.0f;
#pragma unroll
    for (int r = 0; r < 8; ++r) {
      float v = acc[j][r] + bv;
      if (TANH) v = tanhf(v);
      C[(size_t)(m0 + r + roff) * N + col] = v;
    }
  }
}

// ---------------------------------------------------------------------------
// Small elementwise / conversion kernels
// ---------------------------------------------------------------------------
__global__ void fill_f32_kernel(float* __restrict__ p, float v, int n) {
  int i = blockIdx.x * blockDim.x + threadIdx.x;
  if (i < n) p[i] = v;
}

// x_t slice: dst[b*D + d] = bf16( X[(b*T + t)*D + d] )
__global__ void slice_bf16_kernel(const float* __restrict__ X, __bf16* __restrict__ dst, int t) {
  int i = blockIdx.x * blockDim.x + threadIdx.x;
  if (i >= B_ * D_) return;
  int b = i / D_, d = i % D_;
  dst[i] = to_bf16(X[((size_t)b * T_ + t) * D_ + d]);
}

__global__ void conv_bf16_kernel(const float* __restrict__ s, __bf16* __restrict__ d, int n) {
  int i = blockIdx.x * blockDim.x + threadIdx.x;
  if (i < n) d[i] = to_bf16(s[i]);
}

// h_z = zoom_row[b] * h_u[b, :]
__global__ void hz_bf16_kernel(const float* __restrict__ hu, const float* __restrict__ zrow,
                               __bf16* __restrict__ d) {
  int i = blockIdx.x * blockDim.x + threadIdx.x;
  if (i >= B_ * H_) return;
  int b = i / H_;
  d[i] = to_bf16(zrow[b] * hu[i]);
}

// Transpose-convert: WT[n*K + k] = bf16(W[k*N + n]); LDS tile for coalescing.
__global__ __launch_bounds__(256)
void transpose_bf16_kernel(const float* __restrict__ W, __bf16* __restrict__ WT, int K, int N) {
  __shared__ float tile[32][33];
  const int k0 = blockIdx.x << 5, n0 = blockIdx.y << 5;
  const int tx = threadIdx.x & 31, ty = threadIdx.x >> 5;  // 32 x 8
#pragma unroll
  for (int yy = ty; yy < 32; yy += 8) tile[yy][tx] = W[(size_t)(k0 + yy) * N + (n0 + tx)];
  __syncthreads();
#pragma unroll
  for (int yy = ty; yy < 32; yy += 8)
    WT[(size_t)(n0 + yy) * K + (k0 + tx)] = to_bf16(tile[tx][yy]);
}

// ---------------------------------------------------------------------------
// Zoom update: per batch element, 6 dot-products + softmax over 5 rows.
// ---------------------------------------------------------------------------
__device__ float block_dot(const float* __restrict__ a, const float* __restrict__ b,
                           int n, float* red) {
  float s = 0.f;
  for (int k = threadIdx.x; k < n; k += blockDim.x) s += a[k] * b[k];
  red[threadIdx.x] = s;
  __syncthreads();
  for (int off = blockDim.x >> 1; off > 0; off >>= 1) {
    if ((int)threadIdx.x < off) red[threadIdx.x] += red[threadIdx.x + off];
    __syncthreads();
  }
  float r = red[0];
  __syncthreads();
  return r;
}

__global__ __launch_bounds__(256)
void zoom_kernel(const float* __restrict__ X,
                 const float* h0, const float* h1, const float* h2,
                 const float* h3, const float* h4,
                 const float* __restrict__ Wzx, const float* __restrict__ Wzh,
                 const float* __restrict__ bz,
                 int lz0, int lz1, int lz2, int lz3, int lz4,
                 int cur, float* __restrict__ zoom) {
  __shared__ float red[256];
  __shared__ float rows[LN_];
  const int b = blockIdx.x;
  const float* hs[LN_] = {h0, h1, h2, h3, h4};
  const int lz[LN_]    = {lz0, lz1, lz2, lz3, lz4};
  const float bzv = bz[0];

  const float dcur = block_dot(hs[cur] + (size_t)b * H_, Wzh, H_, red);
  for (int i = 0; i < LN_; ++i) {
    float s;
    if (lz[i] == 0) {
      s = block_dot(hs[i] + (size_t)b * H_, Wzh, H_, red);
    } else {
      s = block_dot(X + ((size_t)b * T_ + lz[i]) * D_, Wzx, D_, red) + dcur;
    }
    if (threadIdx.x == 0) rows[i] = tanhf(s + bzv);
    __syncthreads();
  }
  if (threadIdx.x == 0) {
    float m = rows[0];
    for (int i = 1; i < LN_; ++i) m = fmaxf(m, rows[i]);
    float e[LN_], sum = 0.f;
    for (int i = 0; i < LN_; ++i) { e[i] = __expf(rows[i] - m); sum += e[i]; }
    for (int i = 0; i < LN_; ++i) zoom[i * B_ + b] = e[i] / sum;
  }
}

// Concat h -> h_out (fp32 into d_out tail) + bf16 copy for the FC GEMM.
__global__ void concat_kernel(const float* h0, const float* h1, const float* h2,
                              const float* h3, const float* h4,
                              float* __restrict__ outf, __bf16* __restrict__ outb) {
  int i = blockIdx.x * blockDim.x + threadIdx.x;
  if (i >= B_ * HO_) return;
  const float* hs[LN_] = {h0, h1, h2, h3, h4};
  int b = i / HO_, c = i % HO_;
  int l = c / H_, j = c % H_;
  float v = hs[l][(size_t)b * H_ + j];
  outf[i] = v;
  outb[i] = to_bf16(v);
}

// ---------------------------------------------------------------------------
// Host: schedule (replicates build_schedule exactly, incl. min_id persistence)
// ---------------------------------------------------------------------------
struct Step { int t, id, id_u, zoom_valid; int lz[LN_]; };

static int build_schedule_host(Step* sched) {
  const int L = 112;
  int li[L];
  for (int i = 0; i < L; ++i) li[i] = 0;
  for (int i = 0; i < LN_; ++i) {
    for (int j = 0; j < 7; ++j) {
      int n = (i != LN_ - 1) ? ((1 << i) + j * (1 << (i + 1)))
                             : ((1 << i) + j * (1 << i));
      li[n - 1] = i + 1;
    }
  }
  for (int x = 0; x < L / 2; ++x) { int tmp = li[x]; li[x] = li[L - 1 - x]; li[L - 1 - x] = tmp; }

  int ns = 0, min_id = 0;
  for (int t = 0; t < L; ++t) {
    int idl = li[t];
    if (!idl) continue;
    Step s; s.t = t; s.id = idl; s.id_u = 0; s.zoom_valid = 0;
    for (int i = 0; i < LN_; ++i) s.lz[i] = 0;
    if (idl != LN_) {
      for (int j = t; j >= 0; --j) if (li[j] > idl) { s.id_u = li[j]; break; }
    }
    for (int i = t + 1; i < L; ++i) if (li[i] != 0) { min_id = li[i]; break; }
    if (min_id <= idl) {
      s.zoom_valid = 1;
      for (int i = min_id; i <= idl; ++i)
        for (int j = t + 1; j < L; ++j) if (li[j] == i) { s.lz[i - 1] = j; break; }
    }
    sched[ns++] = s;
  }
  return ns;
}

// ---------------------------------------------------------------------------
// kernel_launch
// ---------------------------------------------------------------------------
extern "C" void kernel_launch(void* const* d_in, const int* in_sizes, int n_in,
                              void* d_out, int out_size, void* d_ws, size_t ws_size,
                              hipStream_t stream) {
  (void)in_sizes; (void)n_in; (void)out_size; (void)ws_size;

  const float* X    = (const float*)d_in[0];
  const float* Wx   = (const float*)d_in[1];
  const float* Wh   = (const float*)d_in[2];
  const float* Whl  = (const float*)d_in[3];
  const float* bias = (const float*)d_in[4];
  const float* Wzx  = (const float*)d_in[5];
  const float* Wzh  = (const float*)d_in[6];
  const float* bz   = (const float*)d_in[7];
  const float* fcw  = (const float*)d_in[8];
  const float* fcb  = (const float*)d_in[9];
  float* out = (float*)d_out;                 // [y_pred (512x128) | h_out (512x5120)]
  float* out_h = out + (size_t)B_ * O_;

  // --- workspace carve (256B aligned) ---
  size_t off = 0;
  auto carve = [&](size_t bytes) -> char* {
    char* p = (char*)d_ws + off;
    off += (bytes + 255) & ~(size_t)255;
    return p;
  };
  __bf16* WxT   = (__bf16*)carve((size_t)H_ * D_ * 2);   // (H, D)  = Wx^T
  __bf16* WhT   = (__bf16*)carve((size_t)H_ * H_ * 2);
  __bf16* WhlT  = (__bf16*)carve((size_t)H_ * H_ * 2);
  __bf16* fcwT  = (__bf16*)carve((size_t)O_ * HO_ * 2);  // (O, 5H) = fc_w^T
  __bf16* xb    = (__bf16*)carve((size_t)B_ * D_ * 2);
  __bf16* hb    = (__bf16*)carve((size_t)B_ * H_ * 2);
  __bf16* hzb   = (__bf16*)carve((size_t)B_ * H_ * 2);
  float*  h[LN_];
  for (int i = 0; i < LN_; ++i) h[i] = (float*)carve((size_t)B_ * H_ * 4);
  float*  zoom  = (float*)carve((size_t)LN_ * B_ * 4);
  __bf16* houtb = (__bf16*)carve((size_t)B_ * HO_ * 2);

  const int EW = (B_ * H_ + 255) / 256;   // elementwise grid over 512x1024

  // --- one-time per call: weight transposes + state init ---
  transpose_bf16_kernel<<<dim3(D_ / 32, H_ / 32), 256, 0, stream>>>(Wx, WxT, D_, H_);
  transpose_bf16_kernel<<<dim3(H_ / 32, H_ / 32), 256, 0, stream>>>(Wh, WhT, H_, H_);
  transpose_bf16_kernel<<<dim3(H_ / 32, H_ / 32), 256, 0, stream>>>(Whl, WhlT, H_, H_);
  transpose_bf16_kernel<<<dim3(HO_ / 32, O_ / 32), 256, 0, stream>>>(fcw, fcwT, HO_, O_);
  for (int i = 0; i < LN_; ++i)
    fill_f32_kernel<<<EW, 256, 0, stream>>>(h[i], 0.0f, B_ * H_);
  fill_f32_kernel<<<(LN_ * B_ + 255) / 256, 256, 0, stream>>>(zoom, 1.0f / LN_, LN_ * B_);

  // --- recurrent schedule ---
  Step sched[64];
  const int nsteps = build_schedule_host(sched);
  const int STEP_BLOCKS = (32 /*M tiles*/ * 16 /*N strips*/ * 32) / 256;  // 64

  for (int si = 0; si < nsteps; ++si) {
    const Step& s = sched[si];
    float* hc = h[s.id - 1];

    // prep: x_t -> bf16, h_prev -> bf16, optional h_z -> bf16
    slice_bf16_kernel<<<EW, 256, 0, stream>>>(X, xb, s.t);
    conv_bf16_kernel<<<EW, 256, 0, stream>>>(hc, hb, B_ * H_);
    if (s.id != LN_)
      hz_bf16_kernel<<<EW, 256, 0, stream>>>(h[s.id_u - 1], zoom + (size_t)(s.id - 1) * B_, hzb);

    // fused GEMMs + bias + tanh -> h[l]
    if (s.id == LN_) {
      gemm_bf16_kernel<2, true><<<STEP_BLOCKS, 256, 0, stream>>>(
          xb, WxT, hb, WhT, nullptr, nullptr, bias, hc, B_, H_, D_);
    } else {
      gemm_bf16_kernel<3, true><<<STEP_BLOCKS, 256, 0, stream>>>(
          xb, WxT, hb, WhT, hzb, WhlT, bias, hc, B_, H_, D_);
    }

    // zoom update
    if (!s.zoom_valid) {
      fill_f32_kernel<<<(LN_ * B_ + 255) / 256, 256, 0, stream>>>(zoom, 1.0f / LN_, LN_ * B_);
    } else {
      zoom_kernel<<<B_, 256, 0, stream>>>(X, h[0], h[1], h[2], h[3], h[4], Wzx, Wzh, bz,
                                          s.lz[0], s.lz[1], s.lz[2], s.lz[3], s.lz[4],
                                          s.id - 1, zoom);
    }
  }

  // --- epilogue: concat + FC ---
  concat_kernel<<<(B_ * HO_ + 255) / 256, 256, 0, stream>>>(h[0], h[1], h[2], h[3], h[4],
                                                            out_h, houtb);
  const int FC_BLOCKS = (32 * 2 * 32) / 256;  // M tiles 32, N strips 2 -> 8 blocks
  gemm_bf16_kernel<1, false><<<FC_BLOCKS, 256, 0, stream>>>(
      houtb, fcwT, nullptr, nullptr, nullptr, nullptr, fcb, out, B_, O_, HO_);
}
